// PatchMerging_51015621542354
// MI455X (gfx1250) — matrix operand order repaired and checked by reading, not compiled, not run on hardware
//
#include <hip/hip_runtime.h>
#include <hip/hip_bf16.h>
#include <stdint.h>

// ---------------- problem constants ----------------
#define BB   8
#define CC   96
#define HH   224
#define WW   224
#define NN   (HH * WW)          // 50176
#define H2   112
#define W2   112
#define S2   (H2 * W2)          // 12544 points per batch after merge
#define K4   (4 * CC)           // 384  (LN / GEMM K)
#define C2   (2 * CC)           // 192  (GEMM N)
#define ROWS (BB * S2)          // 100352 GEMM rows
#define KT   (K4 / 32)          // 12 k-steps of 32
#define NT   (C2 / 16)          // 12 n-tiles of 16

// GEMM tiling: 256 threads = 8 waves; wave = 32 rows x 48 cols (2x3 wmma tiles)
#define WG_ROWS 64
#define SMP     392             // LDS row pitch in ushorts (384 + 8 pad, keeps 16B align)

typedef __attribute__((ext_vector_type(16))) __bf16     v16bf;
typedef __attribute__((ext_vector_type(8)))  float      v8f;
typedef __attribute__((ext_vector_type(4)))  unsigned   u32x4;

union Frag16 {
    u32x4          q[2];
    unsigned short us[16];
    v16bf          bf;
};

__device__ __forceinline__ unsigned short f2bf(float f) {
    unsigned u = __float_as_uint(f);
    u += 0x7FFFu + ((u >> 16) & 1u);        // round-to-nearest-even
    return (unsigned short)(u >> 16);
}
__device__ __forceinline__ float bf2f(unsigned bits) {
    return __uint_as_float(bits << 16);
}

// ---------------------------------------------------------------------------
// Kernel 1: scatter.  One thread per point (b,n): reads its 96 channels
// (coalesced across lanes since consecutive lanes take consecutive n), packs
// to bf16 in registers, writes ONE contiguous 192-byte chunk (12 x b128) at
// Xm[b*S2+s][q*C .. q*C+95], plus the quadrant mask Mq[b*S2+s][q].
// ---------------------------------------------------------------------------
__global__ void pm_scatter(const int* __restrict__ pos,
                           const float* __restrict__ feat,
                           const float* __restrict__ mask,
                           unsigned short* __restrict__ Xm,
                           float* __restrict__ Mq) {
    int t = blockIdx.x * blockDim.x + threadIdx.x;   // over BB*NN points
    if (t >= BB * NN) return;
    int n = t % NN;
    int b = t / NN;

    int x = pos[((long long)b * 2 + 0) * NN + n];
    int y = pos[((long long)b * 2 + 1) * NN + n];
    float m = mask[(long long)b * NN + n];

    int s = (y >> 1) * W2 + (x >> 1);
    int q = (y & 1) + ((x & 1) << 1);

    const float* fp = feat + (size_t)b * CC * NN + n;
    unsigned buf[CC / 2];
#pragma unroll
    for (int cp = 0; cp < CC / 2; ++cp) {
        float f0 = fp[(size_t)(2 * cp)     * NN] * m;
        float f1 = fp[(size_t)(2 * cp + 1) * NN] * m;
        buf[cp] = (unsigned)f2bf(f0) | ((unsigned)f2bf(f1) << 16);
    }
    u32x4* dst = (u32x4*)(Xm + ((size_t)(b * S2 + s)) * K4 + q * CC);
    const u32x4* src = (const u32x4*)buf;
#pragma unroll
    for (int j = 0; j < CC / 8; ++j) dst[j] = src[j];

    Mq[(size_t)(b * S2 + s) * 4 + q] = m;
}

// ---------------------------------------------------------------------------
// Kernel 2: pre-swizzle red_w (192 x 384 fp32) into WMMA B-fragment order,
//           bf16.  B-frag (32x16 KxN): lane L holds N = nt*16 + (L&15), halves j
//           hold K = kt*32 + (L>>4)*16 + j.   Wfrag[kt][nt][lane][j]
// ---------------------------------------------------------------------------
__global__ void pm_wswizzle(const float* __restrict__ red_w,
                            unsigned short* __restrict__ Wfrag) {
    int tid = blockIdx.x * blockDim.x + threadIdx.x;
    if (tid >= KT * NT * 32) return;
    int lane = tid & 31;
    int nt   = (tid >> 5) % NT;
    int kt   = (tid >> 5) / NT;
    int k = kt * 32 + (lane >> 4) * 16;
    unsigned short* dst = Wfrag + (size_t)tid * 16;
#pragma unroll
    for (int j = 0; j < 16; ++j) {
        int ch = nt * 16 + (lane & 15);             // output channel (row of red_w)
        dst[j] = f2bf(red_w[(size_t)ch * K4 + k + j]);
    }
}

// ---------------------------------------------------------------------------
// Kernel 3: pos_out (int32 meshgrid) and mask_out (2x2 max -> int32)
// ---------------------------------------------------------------------------
__global__ void pm_posmask(const float* __restrict__ Mq,
                           int* __restrict__ pos_out,
                           int* __restrict__ mask_out) {
    int tid = blockIdx.x * blockDim.x + threadIdx.x;
    if (tid >= BB * S2) return;
    int b = tid / S2;
    int s = tid % S2;
    pos_out[((long long)b * 2 + 0) * S2 + s] = s % W2;
    pos_out[((long long)b * 2 + 1) * S2 + s] = s / W2;
    const float* mq = Mq + (size_t)tid * 4;
    float mv = fmaxf(fmaxf(mq[0], mq[1]), fmaxf(mq[2], mq[3]));
    mask_out[tid] = (int)mv;
}

// ---------------------------------------------------------------------------
// Kernel 4: fused LayerNorm + bf16 WMMA GEMM.
//   Workgroup = 256 threads (8 wave32), 64 rows of Xm, all 192 out channels.
//   Phase 1: load rows, wave-shuffle mean/var, normalize -> bf16 in LDS.
//   Phase 2: wave = 32 rows x 48 cols: every B fragment feeds 2 WMMAs
//            (12 k-steps x 3 n-tiles x 2 m-tiles of v_wmma_f32_16x16x32_bf16)
// ---------------------------------------------------------------------------
__global__ void __launch_bounds__(256)
pm_lngemm(const unsigned short* __restrict__ Xm,
          const unsigned short* __restrict__ Wfrag,
          const float* __restrict__ norm_w,
          const float* __restrict__ norm_b,
          float* __restrict__ out) {
    __shared__ unsigned short sm[WG_ROWS * SMP];     // 50176 B

    const int tid = threadIdx.x;
    const int wg  = blockIdx.x;                      // 0..1567
    const size_t rowBase = (size_t)wg * WG_ROWS;

    // ---------------- Phase 1: LayerNorm into LDS (bf16) ----------------
    {
        const int lrow = tid >> 2;                   // 0..63
        const int part = tid & 3;                    // 0..3 : 96 elems each
        const u32x4* src =
            (const u32x4*)(Xm + (rowBase + lrow) * K4 + part * 96);
        u32x4* dst = (u32x4*)(sm + lrow * SMP + part * 96);

        float sum = 0.f, sq = 0.f;
#pragma unroll
        for (int j = 0; j < 12; ++j) {               // 12 x 16B = 96 ushorts
            u32x4 v = src[j];
            dst[j] = v;                              // stash raw bf16 in LDS
#pragma unroll
            for (int e = 0; e < 4; ++e) {
                float lo = bf2f(v[e] & 0xFFFFu);
                float hi = bf2f(v[e] >> 16);
                sum += lo + hi;
                sq  += lo * lo + hi * hi;
            }
        }
        // reduce across the 4 lanes that share a row (lanes 4r..4r+3)
        sum += __shfl_xor(sum, 1); sq += __shfl_xor(sq, 1);
        sum += __shfl_xor(sum, 2); sq += __shfl_xor(sq, 2);

        const float inv = 1.0f / (float)K4;
        float mean = sum * inv;
        float var  = sq * inv - mean * mean;
        float rs   = rsqrtf(var + 1e-5f);

#pragma unroll
        for (int j = 0; j < 12; ++j) {
            u32x4 v = dst[j];
            u32x4 o;
#pragma unroll
            for (int e = 0; e < 4; ++e) {
                int k = part * 96 + j * 8 + e * 2;
                float lo = (bf2f(v[e] & 0xFFFFu) - mean) * rs * norm_w[k]   + norm_b[k];
                float hi = (bf2f(v[e] >> 16)     - mean) * rs * norm_w[k+1] + norm_b[k+1];
                o[e] = (unsigned)f2bf(lo) | ((unsigned)f2bf(hi) << 16);
            }
            dst[j] = o;
        }
    }
    __syncthreads();

    // ---------------- Phase 2: WMMA GEMM ----------------
    const int lane = tid & 31;
    const int wv   = tid >> 5;           // wave 0..7
    const int wm   = wv & 1;             // row half: 32 rows (2 m-tiles)
    const int wn   = wv >> 1;            // col quarter: 48 cols (3 n-tiles)

    v8f acc[2][3];
#pragma unroll
    for (int mi = 0; mi < 2; ++mi)
#pragma unroll
        for (int nt = 0; nt < 3; ++nt)
            acc[mi][nt] = v8f{0.f,0.f,0.f,0.f,0.f,0.f,0.f,0.f};

    const int arow = wm * 32 + (lane & 15);
    const int ksel = (lane >> 4) * 16;   // halves
    const u32x4* wfq = (const u32x4*)Wfrag;

    for (int kt = 0; kt < KT; ++kt) {
        Frag16 a0, a1;
        const u32x4* ap0 = (const u32x4*)(sm + arow * SMP + kt * 32 + ksel);
        const u32x4* ap1 = (const u32x4*)(sm + (arow + 16) * SMP + kt * 32 + ksel);
        a0.q[0] = ap0[0]; a0.q[1] = ap0[1];
        a1.q[0] = ap1[0]; a1.q[1] = ap1[1];
#pragma unroll
        for (int nt = 0; nt < 3; ++nt) {
            Frag16 bfr;
            size_t bi = ((size_t)(kt * NT + wn * 3 + nt) * 32 + lane) * 2;
            bfr.q[0] = wfq[bi];
            bfr.q[1] = wfq[bi + 1];
            acc[0][nt] = __builtin_amdgcn_wmma_f32_16x16x32_bf16(
                false, a0.bf, false, bfr.bf, (short)0, acc[0][nt], false, false);
            acc[1][nt] = __builtin_amdgcn_wmma_f32_16x16x32_bf16(
                false, a1.bf, false, bfr.bf, (short)0, acc[1][nt], false, false);
        }
    }

    // ---------------- Store: out[b][cch][s] ----------------
    // 12544 % 64 == 0 -> whole workgroup inside one batch image
    const int b   = wg / (S2 / WG_ROWS);
    const int nlo = lane & 15;
#pragma unroll
    for (int mi = 0; mi < 2; ++mi) {
        int sbase = (wg % (S2 / WG_ROWS)) * WG_ROWS + wm * 32 + mi * 16 + ((lane >> 4) << 3);
#pragma unroll
        for (int nt = 0; nt < 3; ++nt) {
            int cch = (wn * 3 + nt) * 16 + nlo;
            float* op = out + ((size_t)b * C2 + cch) * S2 + sbase;
#pragma unroll
            for (int r = 0; r < 8; ++r) op[r] = acc[mi][nt][r];
        }
    }
}

// ---------------------------------------------------------------------------
extern "C" void kernel_launch(void* const* d_in, const int* in_sizes, int n_in,
                              void* d_out, int out_size, void* d_ws, size_t ws_size,
                              hipStream_t stream) {
    const int*   pos    = (const int*)d_in[0];
    const float* feat   = (const float*)d_in[1];
    const float* mask   = (const float*)d_in[2];
    const float* norm_w = (const float*)d_in[3];
    const float* norm_b = (const float*)d_in[4];
    const float* red_w  = (const float*)d_in[5];

    // output tuple, flat 4-byte elements: pos_out(int32) | out(fp32) | mask_out(int32)
    int*   pos_out  = (int*)d_out;
    float* outF     = (float*)d_out + (size_t)BB * 2 * S2;
    int*   mask_out = (int*)((float*)d_out + (size_t)BB * 2 * S2 + (size_t)BB * C2 * S2);

    // workspace layout
    char* ws = (char*)d_ws;
    unsigned short* Xm = (unsigned short*)ws;                         // 77,070,336 B
    size_t off = (size_t)ROWS * K4 * sizeof(unsigned short);
    float* Mq = (float*)(ws + off);                                   // 1,605,632 B
    off += (size_t)ROWS * 4 * sizeof(float);
    unsigned short* Wfrag = (unsigned short*)(ws + off);              // 147,456 B

    // 1) scatter into merged bf16 layout (fully covers Xm/Mq; no init needed)
    {
        int total = BB * NN;                         // one thread per point
        pm_scatter<<<(total + 255) / 256, 256, 0, stream>>>(pos, feat, mask, Xm, Mq);
    }
    // 2) weight swizzle to WMMA B-fragments (bf16, L2 resident)
    {
        int total = KT * NT * 32;
        pm_wswizzle<<<(total + 255) / 256, 256, 0, stream>>>(red_w, Wfrag);
    }
    // 3) pos_out / mask_out
    {
        int total = BB * S2;
        pm_posmask<<<(total + 255) / 256, 256, 0, stream>>>(Mq, pos_out, mask_out);
    }
    // 4) fused LayerNorm + WMMA GEMM
    {
        pm_lngemm<<<ROWS / WG_ROWS, 256, 0, stream>>>(Xm, Wfrag, norm_w, norm_b, outF);
    }
}
